// SpikingSelfAttention_21895743275217
// MI455X (gfx1250) — matrix operand order repaired and checked by reading, not compiled, not run on hardware
//
#include <hip/hip_runtime.h>

// ---------------------------------------------------------------------------
// Spiking self-attention (T=4, B=16, N=1024, D=512, H=8) for gfx1250.
//   pass 1: fused  q/k/v = xf @ W{q,k,v}^T  -> LIF -> kv cumsum -> y  (bf16)
//   pass 2: out = y @ Wp^T + bp                                        (f32)
// BF16 WMMA (v_wmma_f32_16x16x32_bf16) with fp32 accumulation.
// ---------------------------------------------------------------------------

typedef __attribute__((ext_vector_type(8)))  __bf16 v8bf;
typedef __attribute__((ext_vector_type(16))) __bf16 v16bf;
typedef __attribute__((ext_vector_type(8)))  float  v8f;

#define T_STEPS 4
#define MROWS   16384            // B * N
#define DDIM    512
#define LDSS    (DDIM + 8)       // +8 bf16 (16B) pad: shifts banks 4/row

static __device__ __forceinline__ unsigned short f2bf(float f) {
    unsigned int u = __float_as_uint(f);
    u += 0x7FFFu + ((u >> 16) & 1u);           // round-to-nearest-even
    return (unsigned short)(u >> 16);
}

// concatenate two aligned 8x bf16 (16B) loads into a 16-element fragment
static __device__ __forceinline__ v16bf ld_frag(const __bf16* lo, const __bf16* hi) {
    v8bf a = *(const v8bf*)lo;
    v8bf b = *(const v8bf*)hi;
    return __builtin_shufflevector(a, b, 0,1,2,3,4,5,6,7,8,9,10,11,12,13,14,15);
}

static __device__ __forceinline__ v8f wmma_bf16(v16bf a, v16bf b, v8f c) {
    return __builtin_amdgcn_wmma_f32_16x16x32_bf16(false, a, false, b,
                                                   (short)0, c, false, false);
}

// ---------------------------------------------------------------------------
__global__ void f32_to_bf16_vec4(const float* __restrict__ src,
                                 unsigned short* __restrict__ dst, int n4) {
    int i = blockIdx.x * blockDim.x + threadIdx.x;
    if (i < n4) {
        float4 f = ((const float4*)src)[i];
        ushort4 o = make_ushort4(f2bf(f.x), f2bf(f.y), f2bf(f.z), f2bf(f.w));
        ((ushort4*)dst)[i] = o;
    }
}

// ---------------------------------------------------------------------------
// Pass 1: block = (m-tile of 16 rows) x (8 waves x 16 cols = 128 cols).
// grid = (MROWS/16, DDIM/128). 12 accumulators live across the K loop so the
// three weight matrices stream from L2 exactly once per block.
// ---------------------------------------------------------------------------
__global__ __launch_bounds__(256) void snn_qkv_lif(
    const float*          __restrict__ x,    // (T, MROWS, D) fp32
    const unsigned short* __restrict__ wqb,  // (D, D) bf16 row-major
    const unsigned short* __restrict__ wkb,
    const unsigned short* __restrict__ wvb,
    unsigned short*       __restrict__ ybf)  // (T, MROWS, D) bf16
{
    __shared__ __align__(16) unsigned short As[T_STEPS][16][LDSS];

    const int tid  = threadIdx.x;
    const int lane = tid & 31;
    const int wave = tid >> 5;
    const int m0   = blockIdx.x << 4;
    const int n0   = (blockIdx.y << 7) + (wave << 4);

    // Stage x tiles for all 4 timesteps into LDS, fp32 -> bf16 on the fly.
    for (int c = tid; c < T_STEPS * 16 * (DDIM / 4); c += 256) {
        const int t   = c / (16 * (DDIM / 4));
        const int rem = c - t * 16 * (DDIM / 4);
        const int r   = rem / (DDIM / 4);
        const int k4  = (rem - r * (DDIM / 4)) * 4;
        const float4 f = *(const float4*)(x + (((size_t)t * MROWS + m0 + r) * DDIM + k4));
        *(ushort4*)(&As[t][r][k4]) =
            make_ushort4(f2bf(f.x), f2bf(f.y), f2bf(f.z), f2bf(f.w));
    }
    __syncthreads();

    v8f accq[T_STEPS], acck[T_STEPS], accv[T_STEPS];
    const v8f zero = {0.f, 0.f, 0.f, 0.f, 0.f, 0.f, 0.f, 0.f};
#pragma unroll
    for (int t = 0; t < T_STEPS; ++t) { accq[t] = zero; acck[t] = zero; accv[t] = zero; }

    const int r16 = lane & 15;        // A row / B,D column for this lane
    const int hi  = lane >> 4;
    const int kbA = hi << 3;          // A frag: K = kc+kbA..+7  and  kc+16+kbA..+7
    const int kbB = hi << 4;          // unused: B frag takes the same K as A (kbA)

    const __bf16* wq = (const __bf16*)wqb + (size_t)(n0 + r16) * DDIM;
    const __bf16* wk = (const __bf16*)wkb + (size_t)(n0 + r16) * DDIM;
    const __bf16* wv = (const __bf16*)wvb + (size_t)(n0 + r16) * DDIM;

    for (int kc = 0; kc < DDIM; kc += 32) {
        const v16bf bq = ld_frag(wq + kc + kbA, wq + kc + 16 + kbA);
        const v16bf bk = ld_frag(wk + kc + kbA, wk + kc + 16 + kbA);
        const v16bf bv = ld_frag(wv + kc + kbA, wv + kc + 16 + kbA);
#pragma unroll
        for (int t = 0; t < T_STEPS; ++t) {
            const __bf16* ar = (const __bf16*)&As[t][r16][0];
            const v16bf a = ld_frag(ar + kc + kbA, ar + kc + 16 + kbA);
            accq[t] = wmma_bf16(a, bq, accq[t]);
            acck[t] = wmma_bf16(a, bk, acck[t]);
            accv[t] = wmma_bf16(a, bv, accv[t]);
        }
    }

    // Epilogue: LIF recurrence (soft reset), kv cumsum, q*context -> y (bf16).
#pragma unroll
    for (int i = 0; i < 8; ++i) {
        const int row = (hi << 3) + i;
        float mq = 0.f, mk = 0.f, mv = 0.f, ctx = 0.f;
#pragma unroll
        for (int t = 0; t < T_STEPS; ++t) {
            mq = mq * 0.9f + accq[t][i];
            const float sq = (mq - 1.0f >= 0.0f) ? 1.0f : 0.0f;
            mq -= sq;
            mk = mk * 0.9f + acck[t][i];
            const float sk = (mk - 1.0f >= 0.0f) ? 1.0f : 0.0f;
            mk -= sk;
            mv = mv * 0.9f + accv[t][i];
            const float sv = (mv - 1.0f >= 0.0f) ? 1.0f : 0.0f;
            mv -= sv;
            ctx += sk * sv;
            const float yv = sq * ctx;   // exact small integer, bf16-lossless
            ybf[((size_t)t * MROWS + m0 + row) * DDIM + n0 + r16] = f2bf(yv);
        }
    }
}

// ---------------------------------------------------------------------------
// Pass 2: out = y @ Wp^T + bp.  Rows span T*MROWS. grid = (T*MROWS/16, 4).
// ---------------------------------------------------------------------------
__global__ __launch_bounds__(256) void snn_proj(
    const unsigned short* __restrict__ ybf,  // (T*MROWS, D) bf16
    const unsigned short* __restrict__ wpb,  // (D, D) bf16
    const float*          __restrict__ bp,   // (D,) fp32
    float*                __restrict__ out)  // (T*MROWS, D) fp32
{
    __shared__ __align__(16) unsigned short As[16][LDSS];

    const int tid  = threadIdx.x;
    const int lane = tid & 31;
    const int wave = tid >> 5;
    const int m0   = blockIdx.x << 4;
    const int n0   = (blockIdx.y << 7) + (wave << 4);

    for (int c = tid; c < 16 * (DDIM / 8); c += 256) {
        const int r  = c / (DDIM / 8);
        const int k8 = (c - r * (DDIM / 8)) * 8;
        *(uint4*)(&As[r][k8]) = *(const uint4*)(ybf + ((size_t)(m0 + r) * DDIM + k8));
    }
    __syncthreads();

    v8f acc = {0.f, 0.f, 0.f, 0.f, 0.f, 0.f, 0.f, 0.f};

    const int r16 = lane & 15;
    const int hi  = lane >> 4;
    const int kbA = hi << 3;
    const int kbB = hi << 4;

    const __bf16* wp = (const __bf16*)wpb + (size_t)(n0 + r16) * DDIM;

    for (int kc = 0; kc < DDIM; kc += 32) {
        const v16bf b = ld_frag(wp + kc + kbA, wp + kc + 16 + kbA);
        const __bf16* ar = (const __bf16*)&As[r16][0];
        const v16bf a = ld_frag(ar + kc + kbA, ar + kc + 16 + kbA);
        acc = wmma_bf16(a, b, acc);
    }

    const float bias = bp[n0 + r16];
#pragma unroll
    for (int i = 0; i < 8; ++i) {
        const int row = (hi << 3) + i;
        out[((size_t)(m0 + row)) * DDIM + n0 + r16] = acc[i] + bias;
    }
}

// ---------------------------------------------------------------------------
extern "C" void kernel_launch(void* const* d_in, const int* in_sizes, int n_in,
                              void* d_out, int out_size, void* d_ws, size_t ws_size,
                              hipStream_t stream) {
    (void)in_sizes; (void)n_in; (void)out_size; (void)ws_size;

    const float* x  = (const float*)d_in[0];  // (T,B,N,D)
    const float* Wq = (const float*)d_in[1];  // (D,D)
    const float* Wk = (const float*)d_in[2];
    const float* Wv = (const float*)d_in[3];
    const float* Wp = (const float*)d_in[4];
    const float* bp = (const float*)d_in[5];  // (D,)

    // Workspace layout (bytes):
    //   [0, 64MiB)          y  bf16 (T*MROWS*D)
    //   then 4 x 512KiB     Wq/Wk/Wv/Wp bf16
    unsigned char* ws = (unsigned char*)d_ws;
    const size_t Y_BYTES = (size_t)T_STEPS * MROWS * DDIM * 2;  // 67,108,864
    const size_t W_BYTES = (size_t)DDIM * DDIM * 2;             // 524,288
    unsigned short* ybf = (unsigned short*)ws;
    unsigned short* wqb = (unsigned short*)(ws + Y_BYTES);
    unsigned short* wkb = (unsigned short*)(ws + Y_BYTES + 1 * W_BYTES);
    unsigned short* wvb = (unsigned short*)(ws + Y_BYTES + 2 * W_BYTES);
    unsigned short* wpb = (unsigned short*)(ws + Y_BYTES + 3 * W_BYTES);

    // Convert weights fp32 -> bf16 (tiny; L2-resident afterwards).
    const int wN4 = (DDIM * DDIM) / 4;  // 65536
    f32_to_bf16_vec4<<<wN4 / 256, 256, 0, stream>>>(Wq, wqb, wN4);
    f32_to_bf16_vec4<<<wN4 / 256, 256, 0, stream>>>(Wk, wkb, wN4);
    f32_to_bf16_vec4<<<wN4 / 256, 256, 0, stream>>>(Wv, wvb, wN4);
    f32_to_bf16_vec4<<<wN4 / 256, 256, 0, stream>>>(Wp, wpb, wN4);

    // Pass 1: fused QKV projections + LIF + linear-attention carry.
    snn_qkv_lif<<<dim3(MROWS / 16, DDIM / 128), 256, 0, stream>>>(
        x, wqb, wkb, wvb, ybf);

    // Pass 2: output projection + bias.
    snn_proj<<<dim3((T_STEPS * MROWS) / 16, DDIM / 128), 256, 0, stream>>>(
        ybf, wpb, bp, (float*)d_out);
}
